// CausalMultiHeadSelfAttention_68341519614610
// MI455X (gfx1250) — matrix operand order repaired and checked by reading, not compiled, not run on hardware
//
#include <hip/hip_runtime.h>
#include <cstddef>

// ---------------------------------------------------------------------------
// Causal MHA for MI455X (gfx1250), bf16 WMMA everywhere, flash-style softmax.
// B=2, S=2048, D=1024, H=16, DK=64.
// Flash kernel computes transposed tiles (S^T = K Q^T, O^T = V^T P^T) so the
// softmax row reductions are in-lane + one shfl_xor(16) and m/l are scalars.
// ---------------------------------------------------------------------------

typedef __attribute__((ext_vector_type(16))) __bf16        bf16x16;
typedef __attribute__((ext_vector_type(8)))  float         f32x8;
typedef __attribute__((ext_vector_type(4)))  unsigned int  u32x4;
typedef __attribute__((ext_vector_type(8)))  unsigned int  u32x8;

static __device__ __forceinline__ unsigned short f32_to_bf16(float f) {
    union { float f; unsigned u; } v; v.f = f;
    unsigned u = v.u + 0x7FFFu + ((v.u >> 16) & 1u);   // round-to-nearest-even
    return (unsigned short)(u >> 16);
}

// A-fragment, 16x32 bf16 (M x K), row-major source with row stride `ld`.
// Lanes 0-15: rows M=lane, K=0..15; lanes 16-31: rows M=lane-16,
// K=16..31.  Two 16B loads per lane.
static __device__ __forceinline__ bf16x16
load_frag_a(const unsigned short* base, int ld, int kk, int lane) {
    const unsigned short* p =
        base + (size_t)(lane & 15) * ld + kk + ((lane >> 4) & 1) * 16;
    u32x4 a = *(const u32x4*)p;
    u32x4 b = *(const u32x4*)(p + 8);
    u32x8 v;
    v[0]=a[0]; v[1]=a[1]; v[2]=a[2]; v[3]=a[3];
    v[4]=b[0]; v[5]=b[1]; v[6]=b[2]; v[7]=b[3];
    return __builtin_bit_cast(bf16x16, v);
}

// B-fragment, 32x16 bf16 (K x N) where column n is row n of a row-major
// array with stride `ld`.  Lanes 0-15: col=lane, K=0..15; lanes 16-31:
// col=lane-16, K=16..31.  One contiguous 32B load per lane.
static __device__ __forceinline__ bf16x16
load_frag_b(const unsigned short* base, int ld, int kk, int lane) {
    const unsigned short* p =
        base + (size_t)(lane & 15) * ld + kk + ((lane >> 4) & 1) * 16;
    u32x8 v = *(const u32x8*)p;
    return __builtin_bit_cast(bf16x16, v);
}

static __device__ __forceinline__ f32x8
wmma_bf16(bf16x16 a, bf16x16 b, f32x8 c) {
    return __builtin_amdgcn_wmma_f32_16x16x32_bf16(
        false, a, false, b, (short)0, c, false, false);
}

// ---------------------------------------------------------------------------
// Elementwise f32 -> bf16 cast
// ---------------------------------------------------------------------------
__global__ void cast_f32_to_bf16_kernel(const float* __restrict__ in,
                                        unsigned short* __restrict__ out,
                                        int n) {
    for (int i = blockIdx.x * blockDim.x + threadIdx.x; i < n;
         i += gridDim.x * blockDim.x)
        out[i] = f32_to_bf16(in[i]);
}

// ---------------------------------------------------------------------------
// NT GEMM:  C[M,N] (f32) = A[M,K] (bf16, row-major) * Bw[N,K]^T (bf16).
// Block = 256 threads = 8 wave32, tile 128(M) x 128(N); wave tile 32x64
// (waves 4x2): 6 fragment loads feed 8 WMMAs per k-step.
// grid = (N/128, M/128)
// ---------------------------------------------------------------------------
__global__ __launch_bounds__(256)
void gemm_nt_bf16_kernel(const unsigned short* __restrict__ A,
                         const unsigned short* __restrict__ Bw,
                         float* __restrict__ C,
                         int K, int lda, int ldb, int ldc) {
    const int lane = threadIdx.x & 31;
    const int wave = threadIdx.x >> 5;
    const int wm = wave >> 1, wn = wave & 1;
    const int m0 = blockIdx.y * 128 + wm * 32;
    const int n0 = blockIdx.x * 128 + wn * 64;

    f32x8 c0[4] = {{0,0,0,0,0,0,0,0},{0,0,0,0,0,0,0,0},
                   {0,0,0,0,0,0,0,0},{0,0,0,0,0,0,0,0}};
    f32x8 c1[4] = {{0,0,0,0,0,0,0,0},{0,0,0,0,0,0,0,0},
                   {0,0,0,0,0,0,0,0},{0,0,0,0,0,0,0,0}};

    const unsigned short* a0base = A + (size_t)m0 * lda;
    const unsigned short* a1base = a0base + (size_t)16 * lda;
    const unsigned short* bbase[4] = {
        Bw + (size_t)(n0)      * ldb, Bw + (size_t)(n0 + 16) * ldb,
        Bw + (size_t)(n0 + 32) * ldb, Bw + (size_t)(n0 + 48) * ldb };

    for (int kk = 0; kk < K; kk += 32) {
        if (kk + 256 < K) {   // CDNA5 speculative prefetch of upcoming K-tiles
            __builtin_prefetch(a0base    + (size_t)(lane & 15) * lda + kk + 256, 0, 1);
            __builtin_prefetch(bbase[0]  + (size_t)(lane & 15) * ldb + kk + 256, 0, 1);
        }
        bf16x16 a0 = load_frag_a(a0base, lda, kk, lane);
        bf16x16 a1 = load_frag_a(a1base, lda, kk, lane);
#pragma unroll
        for (int ni = 0; ni < 4; ++ni) {
            bf16x16 bfrag = load_frag_b(bbase[ni], ldb, kk, lane);
            c0[ni] = wmma_bf16(a0, bfrag, c0[ni]);
            c1[ni] = wmma_bf16(a1, bfrag, c1[ni]);
        }
    }

    const int half = lane >> 4, nl = lane & 15;
#pragma unroll
    for (int ni = 0; ni < 4; ++ni) {
#pragma unroll
        for (int r = 0; r < 8; ++r) {
            const int rm = m0 + r + 8 * half;
            C[(size_t)rm        * ldc + n0 + 16 * ni + nl] = c0[ni][r];
            C[(size_t)(rm + 16) * ldc + n0 + 16 * ni + nl] = c1[ni][r];
        }
    }
}

// ---------------------------------------------------------------------------
// RoPE + cast + reorder: P f32 [B,S,D] -> out bf16 [B,H,S,DK].
// One thread per (even,odd) feature pair.  total threads = B*S*H*DK/2 = 2^21.
// ---------------------------------------------------------------------------
__global__ void rope_cast_kernel(const float* __restrict__ P,
                                 const int* __restrict__ tpos,
                                 unsigned short* __restrict__ out) {
    constexpr int S = 2048, Dm = 1024, DK = 64;
    const int idx = blockIdx.x * blockDim.x + threadIdx.x;
    if (idx >= (1 << 21)) return;
    const int j = idx & 31;            // pair index within head (DK/2 = 32)
    const int h = (idx >> 5) & 15;
    const int s = (idx >> 9) & 2047;
    const int b = idx >> 20;

    const float pos = (float)tpos[b * S + s];
    const float inv = __powf(10000.0f, -(float)(2 * j) * (1.0f / 64.0f));
    float sn, cs;
    __sincosf(pos * inv, &sn, &cs);

    const float* row = P + ((size_t)(b * S + s)) * Dm + h * DK;
    const float x1 = row[2 * j], x2 = row[2 * j + 1];

    unsigned short* o = out + (((size_t)(b * 16 + h)) * S + s) * DK;
    o[2 * j]     = f32_to_bf16(cs * x1 - sn * x2);
    o[2 * j + 1] = f32_to_bf16(sn * x1 + cs * x2);
}

// ---------------------------------------------------------------------------
// V transpose + cast: P f32 [B,S,D] -> Vt bf16 [B,H,DK,S]
// ---------------------------------------------------------------------------
__global__ void v_transpose_cast_kernel(const float* __restrict__ P,
                                        unsigned short* __restrict__ vt) {
    constexpr int S = 2048, Dm = 1024, DK = 64;
    const int idx = blockIdx.x * blockDim.x + threadIdx.x;
    if (idx >= (1 << 22)) return;
    const int dk = idx & 63;
    const int h  = (idx >> 6) & 15;
    const int s  = (idx >> 10) & 2047;
    const int b  = idx >> 21;
    const float v = P[((size_t)(b * S + s)) * Dm + h * DK + dk];
    vt[(((size_t)(b * 16 + h)) * DK + dk) * S + s] = f32_to_bf16(v);
}

// ---------------------------------------------------------------------------
// Flash attention (transposed tiles).  grid = (S/128, B*H), block = 256.
// Wave w owns q-rows [q0+16w, q0+16w+16).  S^T = K*Q^T puts each query's
// keys in-lane (8 slots x 2 lane-halves), so softmax stats are per-lane
// scalars with a single shfl_xor(16).  O^T = V^T * P^T accumulates the
// output transposed; P^T round-trips through a per-wave LDS slice to become
// a B-fragment.  Per-wave causal bound -> no block barriers.
// ---------------------------------------------------------------------------
__global__ __launch_bounds__(256)
void flash_attn_kernel(const unsigned short* __restrict__ Q,   // [B,H,S,DK]
                       const unsigned short* __restrict__ Kb,  // [B,H,S,DK]
                       const unsigned short* __restrict__ Vt,  // [B,H,DK,S]
                       unsigned short* __restrict__ Out) {     // [B,S,D] bf16
    constexpr int S = 2048, Dm = 1024, DK = 64;
    constexpr float scale = 0.125f;                 // 1/sqrt(64)

    __shared__ __align__(32) unsigned short Pq[8][16][32];   // [wave][q][key]

    const int lane = threadIdx.x & 31;
    const int wave = threadIdx.x >> 5;
    const int nl   = lane & 15;
    const int half = lane >> 4;

    const int bh = blockIdx.y;
    const int b  = bh >> 4;
    const int h  = bh & 15;
    const int qw = blockIdx.x * 128 + wave * 16;    // first q-row of this wave
    const int qcol = qw + nl;                       // this lane's query row

    // Q rows as B-fragments (col = query, k = dk)
    const unsigned short* qbase = Q + ((size_t)bh * S + qw) * DK;
    const bf16x16 qb0 = load_frag_b(qbase, DK, 0,  lane);
    const bf16x16 qb1 = load_frag_b(qbase, DK, 32, lane);

    // O^T accumulators: rows = features (4 chunks of 16), cols = queries
    f32x8 o[4] = {{0,0,0,0,0,0,0,0},{0,0,0,0,0,0,0,0},
                  {0,0,0,0,0,0,0,0},{0,0,0,0,0,0,0,0}};
    float m_i = -INFINITY, l_i = 0.0f;

    const unsigned short* vtb = Vt + (size_t)bh * DK * S;

    for (int kb = 0; kb < qw + 16; kb += 32) {
        // ---- S^T = K * Q^T : two 16(key) x 16(query) tiles over 32 keys ----
        const unsigned short* kbase = Kb + ((size_t)bh * S + kb) * DK;
        bf16x16 ka0 = load_frag_a(kbase,           DK, 0,  lane);
        bf16x16 ka1 = load_frag_a(kbase,           DK, 32, lane);
        bf16x16 ka2 = load_frag_a(kbase + 16 * DK, DK, 0,  lane);
        bf16x16 ka3 = load_frag_a(kbase + 16 * DK, DK, 32, lane);

        f32x8 st0 = {0,0,0,0,0,0,0,0};
        st0 = wmma_bf16(ka0, qb0, st0);
        st0 = wmma_bf16(ka1, qb1, st0);
        f32x8 st1 = {0,0,0,0,0,0,0,0};
        st1 = wmma_bf16(ka2, qb0, st1);
        st1 = wmma_bf16(ka3, qb1, st1);

        // ---- scale + causal mask (skip mask on interior chunks) ----
        const bool full = (kb + 31 <= qw);          // wave-uniform
        float mx = -INFINITY;
#pragma unroll
        for (int r = 0; r < 8; ++r) {
            float v0 = st0[r] * scale;
            float v1 = st1[r] * scale;
            if (!full) {
                const int k0 = kb + r + 8 * half;   // key of st0[r]
                v0 = (k0      <= qcol) ? v0 : -INFINITY;
                v1 = (k0 + 16 <= qcol) ? v1 : -INFINITY;
            }
            st0[r] = v0; st1[r] = v1;
            mx = fmaxf(mx, fmaxf(v0, v1));
        }
        mx = fmaxf(mx, __shfl_xor(mx, 16, 32));     // combine lane halves

        const float mnew  = fmaxf(m_i, mx);
        const float alpha = __expf(m_i - mnew);
        m_i = mnew;

        float rs = 0.0f;
#pragma unroll
        for (int r = 0; r < 8; ++r) {
            const float p0 = __expf(st0[r] - mnew);
            const float p1 = __expf(st1[r] - mnew);
            rs += p0 + p1;
            Pq[wave][nl][r + 8 * half]      = f32_to_bf16(p0);  // key kb+r+8h
            Pq[wave][nl][16 + r + 8 * half] = f32_to_bf16(p1);
        }
        rs += __shfl_xor(rs, 16, 32);
        l_i = l_i * alpha + rs;

#pragma unroll
        for (int ni = 0; ni < 4; ++ni) {
            o[ni][0] *= alpha; o[ni][1] *= alpha; o[ni][2] *= alpha;
            o[ni][3] *= alpha; o[ni][4] *= alpha; o[ni][5] *= alpha;
            o[ni][6] *= alpha; o[ni][7] *= alpha;
        }

        __threadfence_block();   // order DS store -> DS load within the wave

        // ---- O^T += V^T * P^T ----
        const bf16x16 pb = load_frag_b(&Pq[wave][0][0], 32, 0, lane);
#pragma unroll
        for (int ni = 0; ni < 4; ++ni) {
            bf16x16 va = load_frag_a(vtb + (size_t)(16 * ni) * S, S, kb, lane);
            o[ni] = wmma_bf16(va, pb, o[ni]);
        }
    }

    // ---- normalize and merge heads: Out[b, q, h*64 + feature] ----
    const float rl = 1.0f / l_i;
    unsigned short* orow = Out + ((size_t)(b * S + qcol)) * Dm + h * DK;
#pragma unroll
    for (int ni = 0; ni < 4; ++ni) {
#pragma unroll
        for (int r = 0; r < 8; ++r)
            orow[16 * ni + r + 8 * half] = f32_to_bf16(o[ni][r] * rl);
    }
}

// ---------------------------------------------------------------------------
// Host launch
// ---------------------------------------------------------------------------
extern "C" void kernel_launch(void* const* d_in, const int* in_sizes, int n_in,
                              void* d_out, int out_size, void* d_ws, size_t ws_size,
                              hipStream_t stream) {
    (void)in_sizes; (void)n_in; (void)out_size; (void)ws_size;

    constexpr int Bz = 2, S = 2048, Dm = 1024, H = 16;
    constexpr int M  = Bz * S;                    // 4096 rows

    const float* x  = (const float*)d_in[0];
    const float* Wq = (const float*)d_in[1];
    const float* Wk = (const float*)d_in[2];
    const float* Wv = (const float*)d_in[3];
    const float* Wo = (const float*)d_in[4];
    const int*   tp = (const int*)d_in[5];
    float* out = (float*)d_out;

    char* ws = (char*)d_ws;
    size_t off = 0;
    auto alloc = [&](size_t bytes) -> void* {
        off = (off + 255) & ~(size_t)255;
        void* p = ws + off;
        off += bytes;
        return p;
    };

    unsigned short* xb    = (unsigned short*)alloc((size_t)M * Dm * 2);
    unsigned short* wqb   = (unsigned short*)alloc((size_t)Dm * Dm * 2);
    unsigned short* wkb   = (unsigned short*)alloc((size_t)Dm * Dm * 2);
    unsigned short* wvb   = (unsigned short*)alloc((size_t)Dm * Dm * 2);
    unsigned short* wob   = (unsigned short*)alloc((size_t)Dm * Dm * 2);
    float*          projf = (float*)alloc((size_t)M * Dm * 4);   // reused Q/K/V
    unsigned short* qh    = (unsigned short*)alloc((size_t)M * Dm * 2);
    unsigned short* kh    = (unsigned short*)alloc((size_t)M * Dm * 2);
    unsigned short* vt    = (unsigned short*)alloc((size_t)M * Dm * 2);
    unsigned short* attnb = (unsigned short*)alloc((size_t)M * Dm * 2);

    // 1) cast inputs to bf16
    cast_f32_to_bf16_kernel<<<2048, 256, 0, stream>>>(x,  xb,  M * Dm);
    cast_f32_to_bf16_kernel<<<1024, 256, 0, stream>>>(Wq, wqb, Dm * Dm);
    cast_f32_to_bf16_kernel<<<1024, 256, 0, stream>>>(Wk, wkb, Dm * Dm);
    cast_f32_to_bf16_kernel<<<1024, 256, 0, stream>>>(Wv, wvb, Dm * Dm);
    cast_f32_to_bf16_kernel<<<1024, 256, 0, stream>>>(Wo, wob, Dm * Dm);

    const dim3 ggrid(Dm / 128, M / 128);          // (8, 32)

    // 2) Q = x Wq^T -> RoPE -> [B,H,S,DK] bf16
    gemm_nt_bf16_kernel<<<ggrid, 256, 0, stream>>>(xb, wqb, projf, Dm, Dm, Dm, Dm);
    rope_cast_kernel<<<(M * Dm / 2) / 256, 256, 0, stream>>>(projf, tp, qh);

    // 3) K likewise
    gemm_nt_bf16_kernel<<<ggrid, 256, 0, stream>>>(xb, wkb, projf, Dm, Dm, Dm, Dm);
    rope_cast_kernel<<<(M * Dm / 2) / 256, 256, 0, stream>>>(projf, tp, kh);

    // 4) V -> transposed [B,H,DK,S] bf16
    gemm_nt_bf16_kernel<<<ggrid, 256, 0, stream>>>(xb, wvb, projf, Dm, Dm, Dm, Dm);
    v_transpose_cast_kernel<<<(M * Dm) / 256, 256, 0, stream>>>(projf, vt);

    // 5) flash attention -> attnb bf16 [B,S,D]
    flash_attn_kernel<<<dim3(S / 128, Bz * H), 256, 0, stream>>>(qh, kh, vt, attnb);

    // 6) out = attn @ Wo^T (f32 into d_out)
    gemm_nt_bf16_kernel<<<ggrid, 256, 0, stream>>>(attnb, wob, out, Dm, Dm, Dm, Dm);
}
